// Qwen2MoeExpert_53042846105776
// MI455X (gfx1250) — matrix operand order repaired and checked by reading, not compiled, not run on hardware
//
#include <hip/hip_runtime.h>
#include <hip/hip_bf16.h>

typedef __attribute__((ext_vector_type(16))) _Float16 v16h;
typedef __attribute__((ext_vector_type(8)))  _Float16 v8h;
typedef __attribute__((ext_vector_type(4)))  _Float16 v4h;
typedef __attribute__((ext_vector_type(8)))  float    v8f;
typedef __attribute__((ext_vector_type(4)))  int      i32x4;

#define GROUPQ 128
#define HDIM   2048
#define IDIM   1408
#define TDIM   16384

#define BM 256
#define BN 64
#define BK 32
#define XK 40   // LDS row stride (halves) for X tile: 80B, 16B aligned
#define WK 40   // LDS row stride (halves) for transposed weight tile

// ---- gfx1250 async global->LDS path (guarded; falls back to regs->LDS) -----
#if defined(__has_builtin)
#  if __has_builtin(__builtin_amdgcn_global_load_async_to_lds_b128)
#    define HAVE_ASYNC_LDS 1
#  endif
#  if __has_builtin(__builtin_amdgcn_s_wait_asynccnt)
#    define HAVE_WAIT_ASYNC_BUILTIN 1
#  endif
#endif
#ifndef HAVE_ASYNC_LDS
#  define HAVE_ASYNC_LDS 0
#endif
#ifndef HAVE_WAIT_ASYNC_BUILTIN
#  define HAVE_WAIT_ASYNC_BUILTIN 0
#endif

typedef __attribute__((address_space(1))) i32x4 gi32x4;   // global int4
typedef __attribute__((address_space(3))) i32x4 li32x4;   // LDS int4

__device__ __forceinline__ void wait_async0() {
#if HAVE_ASYNC_LDS
#  if HAVE_WAIT_ASYNC_BUILTIN
    __builtin_amdgcn_s_wait_asynccnt(0);
#  else
    asm volatile("s_wait_asynccnt 0x0" ::: "memory");
#  endif
#endif
}

// copy 64B per thread: global src -> LDS dst (both 16B aligned)
__device__ __forceinline__ void copy64_to_lds(const _Float16* src, _Float16* dst) {
#if HAVE_ASYNC_LDS
    gi32x4* g = (gi32x4*)(src);
    li32x4* l = (li32x4*)(dst);
    __builtin_amdgcn_global_load_async_to_lds_b128(g, l, 0, 0);
    __builtin_amdgcn_global_load_async_to_lds_b128(g, l, 16, 0);
    __builtin_amdgcn_global_load_async_to_lds_b128(g, l, 32, 0);
    __builtin_amdgcn_global_load_async_to_lds_b128(g, l, 48, 0);
#else
    v8h a0 = *(const v8h*)(src);
    v8h a1 = *(const v8h*)(src + 8);
    v8h a2 = *(const v8h*)(src + 16);
    v8h a3 = *(const v8h*)(src + 24);
    *(v8h*)(dst)      = a0;
    *(v8h*)(dst + 8)  = a1;
    *(v8h*)(dst + 16) = a2;
    *(v8h*)(dst + 24) = a3;
#endif
}

// ---- fragment helpers -------------------------------------------------------

__device__ __forceinline__ v16h concat8(v8h lo, v8h hi) {
    return __builtin_shufflevector(lo, hi,
        0, 1, 2, 3, 4, 5, 6, 7, 8, 9, 10, 11, 12, 13, 14, 15);
}

// A-matrix 16x32 f16 fragment: lane m = lane%16, k-half = 8*(lane/16)
__device__ __forceinline__ v16h load_frag_a(const _Float16* p) {
    v8h lo = *(const v8h*)(p);
    v8h hi = *(const v8h*)(p + 16);
    return concat8(lo, hi);
}

// B-matrix 32x16 f16 fragment from K-contiguous (transposed) LDS tile
__device__ __forceinline__ v16h load_frag_b(const _Float16* p) {
    v8h lo = *(const v8h*)(p);
    v8h hi = *(const v8h*)(p + 16);
    return concat8(lo, hi);
}

__device__ __forceinline__ v8f wmma_f16(v16h a, v16h b, v8f c) {
    return __builtin_amdgcn_wmma_f32_16x16x32_f16(
        false, a, false, b, (short)0, c, false, false);
}

// ---- kernel 1: fp32 -> fp16 conversion of activations ----------------------

__global__ __launch_bounds__(256)
void cvt_f32_to_f16_kernel(const float* __restrict__ x,
                           _Float16* __restrict__ xh) {
    const size_t i = ((size_t)blockIdx.x * blockDim.x + threadIdx.x) * 4;
    float4 v = *(const float4*)(x + i);
    v4h o;
    o[0] = (_Float16)v.x; o[1] = (_Float16)v.y;
    o[2] = (_Float16)v.z; o[3] = (_Float16)v.w;
    *(v4h*)(xh + i) = o;
}

// ---- kernel 2: fused gate+up GEMM + GPTQ dequant + SiLU --------------------
// Block tile 256x64, K-step 32, ping-pong LDS, 8 waves each owning 32x64.

__global__ __launch_bounds__(256)
void gateup_kernel(const _Float16* __restrict__ xh,
                   const int*   __restrict__ qwg,
                   const int*   __restrict__ qzg,
                   const float* __restrict__ scg,
                   const int*   __restrict__ qwu,
                   const int*   __restrict__ qzu,
                   const float* __restrict__ scu,
                   _Float16* __restrict__ hbuf) {
    __shared__ _Float16 Xs[2][BM * XK];
    __shared__ _Float16 Wgs[2][BN * WK];   // transposed: [n][k]
    __shared__ _Float16 Wus[2][BN * WK];

    const int t      = threadIdx.x;
    const int lane   = t & 31;
    const int wave_m = t >> 5;              // 0..7 -> 32-row slab
    const int m0     = blockIdx.y * BM;
    const int n0     = blockIdx.x * BN;

    const int dn  = t & 63;
    const int dkc = t >> 6;
    const int col = n0 + dn;
    const int zsh = 4 * (col & 7);

    const int am = lane & 15;
    const int ak = (lane >> 4) << 3;
    const int bn = lane & 15;
    const int bk = (lane >> 4) << 3;

    v8f accg[2][4] = {};
    v8f accu[2][4] = {};

    int qg, qu;
    float sg, su, nzg, nzu;

    auto stage_x = [&](int buf, int k0) {
        copy64_to_lds(xh + (size_t)(m0 + t) * HDIM + k0, &Xs[buf][t * XK]);
    };

    auto load_w = [&](int k0) {
        const int g    = k0 >> 7;
        const int qrow = (k0 >> 3) + dkc;
        qg = qwg[(size_t)qrow * IDIM + col];
        qu = qwu[(size_t)qrow * IDIM + col];
        sg = scg[(size_t)g * IDIM + col];
        su = scu[(size_t)g * IDIM + col];
        const float zg = (float)(((qzg[(size_t)g * (IDIM / 8) + (col >> 3)] >> zsh) & 0xF) + 1);
        const float zu = (float)(((qzu[(size_t)g * (IDIM / 8) + (col >> 3)] >> zsh) & 0xF) + 1);
        nzg = -zg * sg;
        nzu = -zu * su;
    };

    auto commit_w = [&](int buf) {
        v8h wg, wu;
#pragma unroll
        for (int j = 0; j < 8; ++j) {
            wg[j] = (_Float16)fmaf((float)((qg >> (4 * j)) & 0xF), sg, nzg);
            wu[j] = (_Float16)fmaf((float)((qu >> (4 * j)) & 0xF), su, nzu);
        }
        *(v8h*)(&Wgs[buf][dn * WK + dkc * 8]) = wg;
        *(v8h*)(&Wus[buf][dn * WK + dkc * 8]) = wu;
    };

    // compute with B-fragment double buffering: LDS latency hidden by 4 WMMAs
    auto compute = [&](int buf) {
        v16h a0 = load_frag_a(&Xs[buf][(wave_m * 32 + am) * XK + ak]);
        v16h a1 = load_frag_a(&Xs[buf][(wave_m * 32 + 16 + am) * XK + ak]);
        v16h bg = load_frag_b(&Wgs[buf][bn * WK + bk]);
        v16h bu = load_frag_b(&Wus[buf][bn * WK + bk]);
#pragma unroll
        for (int wn = 0; wn < 4; ++wn) {
            v16h bgn, bun;
            if (wn < 3) {
                const int nrow = ((wn + 1) * 16 + bn) * WK + bk;
                bgn = load_frag_b(&Wgs[buf][nrow]);
                bun = load_frag_b(&Wus[buf][nrow]);
            }
            accg[0][wn] = wmma_f16(a0, bg, accg[0][wn]);
            accg[1][wn] = wmma_f16(a1, bg, accg[1][wn]);
            accu[0][wn] = wmma_f16(a0, bu, accu[0][wn]);
            accu[1][wn] = wmma_f16(a1, bu, accu[1][wn]);
            bg = bgn;
            bu = bun;
        }
    };

    // prologue
    stage_x(0, 0);
    load_w(0);
    commit_w(0);
    wait_async0();
    __syncthreads();

    int i = 0;
    for (int k0 = 0; k0 < HDIM; k0 += BK, ++i) {
        const int  cur     = i & 1;
        const bool hasNext = (k0 + BK) < HDIM;
        if (hasNext) {
            stage_x(1 - cur, k0 + BK);   // async copies fly over WMMAs
            load_w(k0 + BK);
        }
        compute(cur);
        if (hasNext) {
            commit_w(1 - cur);
            wait_async0();
            __syncthreads();
        }
    }

    // epilogue: h = silu(g) * u, rcp-based sigmoid, store f16
    const int cn  = lane & 15;
    const int crb = (lane >> 4) << 3;
#pragma unroll
    for (int wm = 0; wm < 2; ++wm) {
#pragma unroll
        for (int wn = 0; wn < 4; ++wn) {
#pragma unroll
            for (int r = 0; r < 8; ++r) {
                const float gv  = accg[wm][wn][r];
                const float uv  = accu[wm][wn][r];
                const float sig = __builtin_amdgcn_rcpf(1.0f + __expf(-gv));
                const int row   = m0 + wave_m * 32 + wm * 16 + crb + r;
                const int c     = n0 + wn * 16 + cn;
                hbuf[(size_t)row * IDIM + c] = (_Float16)(gv * sig * uv);
            }
        }
    }
}

// ---- kernel 3: down GEMM with GPTQ dequant, f32 output ---------------------

__global__ __launch_bounds__(256)
void down_kernel(const _Float16* __restrict__ hbuf,
                 const int*   __restrict__ qw,
                 const int*   __restrict__ qz,
                 const float* __restrict__ sc,
                 float* __restrict__ out) {
    __shared__ _Float16 Xs[2][BM * XK];
    __shared__ _Float16 Ws[2][BN * WK];

    const int t      = threadIdx.x;
    const int lane   = t & 31;
    const int wave_m = t >> 5;
    const int m0     = blockIdx.y * BM;
    const int n0     = blockIdx.x * BN;

    const int dn  = t & 63;
    const int dkc = t >> 6;
    const int col = n0 + dn;
    const int zsh = 4 * (col & 7);

    const int am = lane & 15;
    const int ak = (lane >> 4) << 3;
    const int bn = lane & 15;
    const int bk = (lane >> 4) << 3;

    v8f acc[2][4] = {};

    int q;
    float s, nz;

    auto stage_x = [&](int buf, int k0) {
        copy64_to_lds(hbuf + (size_t)(m0 + t) * IDIM + k0, &Xs[buf][t * XK]);
    };

    auto load_w = [&](int k0) {
        const int g    = k0 >> 7;
        const int qrow = (k0 >> 3) + dkc;
        q = qw[(size_t)qrow * HDIM + col];
        s = sc[(size_t)g * HDIM + col];
        const float z = (float)(((qz[(size_t)g * (HDIM / 8) + (col >> 3)] >> zsh) & 0xF) + 1);
        nz = -z * s;
    };

    auto commit_w = [&](int buf) {
        v8h w;
#pragma unroll
        for (int j = 0; j < 8; ++j) {
            w[j] = (_Float16)fmaf((float)((q >> (4 * j)) & 0xF), s, nz);
        }
        *(v8h*)(&Ws[buf][dn * WK + dkc * 8]) = w;
    };

    auto compute = [&](int buf) {
        v16h a0 = load_frag_a(&Xs[buf][(wave_m * 32 + am) * XK + ak]);
        v16h a1 = load_frag_a(&Xs[buf][(wave_m * 32 + 16 + am) * XK + ak]);
        v16h b  = load_frag_b(&Ws[buf][bn * WK + bk]);
#pragma unroll
        for (int wn = 0; wn < 4; ++wn) {
            v16h bnx;
            if (wn < 3) {
                bnx = load_frag_b(&Ws[buf][((wn + 1) * 16 + bn) * WK + bk]);
            }
            acc[0][wn] = wmma_f16(a0, b, acc[0][wn]);
            acc[1][wn] = wmma_f16(a1, b, acc[1][wn]);
            b = bnx;
        }
    };

    stage_x(0, 0);
    load_w(0);
    commit_w(0);
    wait_async0();
    __syncthreads();

    int i = 0;
    for (int k0 = 0; k0 < IDIM; k0 += BK, ++i) {
        const int  cur     = i & 1;
        const bool hasNext = (k0 + BK) < IDIM;
        if (hasNext) {
            stage_x(1 - cur, k0 + BK);
            load_w(k0 + BK);
        }
        compute(cur);
        if (hasNext) {
            commit_w(1 - cur);
            wait_async0();
            __syncthreads();
        }
    }

    const int cn  = lane & 15;
    const int crb = (lane >> 4) << 3;
#pragma unroll
    for (int wm = 0; wm < 2; ++wm) {
#pragma unroll
        for (int wn = 0; wn < 4; ++wn) {
#pragma unroll
            for (int r = 0; r < 8; ++r) {
                const int row = m0 + wave_m * 32 + wm * 16 + crb + r;
                const int c   = n0 + wn * 16 + cn;
                out[(size_t)row * HDIM + c] = acc[wm][wn][r];
            }
        }
    }
}

// ---- launcher ---------------------------------------------------------------

extern "C" void kernel_launch(void* const* d_in, const int* in_sizes, int n_in,
                              void* d_out, int out_size, void* d_ws, size_t ws_size,
                              hipStream_t stream) {
    (void)in_sizes; (void)n_in; (void)out_size; (void)ws_size;

    const float* x   = (const float*)d_in[0];
    const int*   qwg = (const int*)d_in[1];
    const int*   qzg = (const int*)d_in[2];
    const float* scg = (const float*)d_in[3];
    const int*   qwu = (const int*)d_in[4];
    const int*   qzu = (const int*)d_in[5];
    const float* scu = (const float*)d_in[6];
    const int*   qwd = (const int*)d_in[7];
    const int*   qzd = (const int*)d_in[8];
    const float* scd = (const float*)d_in[9];
    float* out = (float*)d_out;

    // workspace: xh (T*H f16, 64 MiB) | hbuf (T*I f16, 44 MiB)
    _Float16* xh   = (_Float16*)d_ws;
    _Float16* hbuf = xh + (size_t)TDIM * HDIM;

    {
        const size_t total = (size_t)TDIM * HDIM;          // /1024 exact
        cvt_f32_to_f16_kernel<<<(unsigned)(total / (256 * 4)), 256, 0, stream>>>(x, xh);
    }
    {
        dim3 grid(IDIM / BN, TDIM / BM);                   // (22, 64)
        gateup_kernel<<<grid, 256, 0, stream>>>(xh, qwg, qzg, scg,
                                                qwu, qzu, scu, hbuf);
    }
    {
        dim3 grid(HDIM / BN, TDIM / BM);                   // (32, 64)
        down_kernel<<<grid, 256, 0, stream>>>(hbuf, qwd, qzd, scd, out);
    }
}
